// MultiHeadSelfAttention_8727373545563
// MI455X (gfx1250) — matrix operand order repaired and checked by reading, not compiled, not run on hardware
//
#include <hip/hip_runtime.h>

// MI455X / gfx1250 multi-head self-attention, bf16 WMMA pipeline.
// B=4, S=2048, D=1024, H=16, DK=64.  All matmuls via v_wmma_f32_16x16x32_bf16.
// Attention stages K / V^T key-blocks into LDS with
// global_load_async_to_lds_b128 (ASYNCcnt), double-buffered across the
// 4 waves of a workgroup (which share the same (b,h) and key stream).

typedef __attribute__((ext_vector_type(16))) __bf16 v16bf;
typedef __attribute__((ext_vector_type(8)))  __bf16 v8bf;
typedef __attribute__((ext_vector_type(8)))  float  v8f;

constexpr int Bq  = 4;
constexpr int Sq  = 2048;
constexpr int Dq  = 1024;
constexpr int Hq  = 16;
constexpr int DKq = 64;
constexpr int NT  = Bq * Sq;   // 8192 tokens

// LDS staging strides (bf16 elements), padded to de-conflict banks while
// keeping 16B alignment of every fragment chunk.
constexpr int KST = 72;   // K block: 32 rows x 64 el
constexpr int VST = 40;   // V^T block: 64 rows x 32 el
constexpr int KEL = 32 * KST;       // 2304 el
constexpr int VEL = 64 * VST;       // 2560 el
constexpr int BUFEL = KEL + VEL;    // 4864 el per buffer

__device__ __forceinline__ v8f wmma_bf16(v16bf a, v16bf b, v8f c) {
  return __builtin_amdgcn_wmma_f32_16x16x32_bf16(false, a, false, b, (short)0, c,
                                                 false, false);
}

__device__ __forceinline__ v16bf pack16(v8bf lo, v8bf hi) {
  v16bf r;
#pragma unroll
  for (int i = 0; i < 8; ++i) { r[i] = lo[i]; r[i + 8] = hi[i]; }
  return r;
}

// A fragment (16x32 bf16) from a row-major matrix with row stride `stride`:
// lane holds m = lane&15; k = 16*(lane>>4) + 0..15 (contiguous).
__device__ __forceinline__ v16bf load_afrag(const __bf16* row, int ks, int half) {
  v8bf lo = *(const v8bf*)(row + ks + 16 * half);
  v8bf hi = *(const v8bf*)(row + ks + 16 * half + 8);
  return pack16(lo, hi);
}

// ---- async global -> LDS copy (16 bytes per lane), tracked by ASYNCcnt ----
__device__ __forceinline__ void async_ld16(const __bf16* lds_dst,
                                           const __bf16* gsrc) {
  // Generic pointers to LDS carry the LDS offset in their low 32 bits
  // (ISA 10.2).
  unsigned loff = (unsigned)(unsigned long long)lds_dst;
  asm volatile("global_load_async_to_lds_b128 %0, %1, off"
               :: "v"(loff), "v"(gsrc)
               : "memory");
}

__device__ __forceinline__ void wait_async0() {
#if __has_builtin(__builtin_amdgcn_s_wait_asynccnt)
  __builtin_amdgcn_s_wait_asynccnt(0);
#else
  asm volatile("s_wait_asynccnt 0" ::: "memory");
#endif
}

// ---------------- kernel 1: f32 -> bf16 conversion ----------------
__global__ void cvt_f32_bf16(const float* __restrict__ in, __bf16* __restrict__ out,
                             int n) {
  int i = blockIdx.x * blockDim.x + threadIdx.x;
  int stride = gridDim.x * blockDim.x;
  for (; i < n; i += stride) out[i] = (__bf16)in[i];
}

// ---------------- kernel 2: fused QKV projection ----------------
// y = x @ W^T.  Wave computes 32(M tokens) x 16(N outputs) for q,k,v:
// 6 WMMAs per K-step, weight B-fragments reused across both M-subtiles.
// q,k written [B,H,S,DK]; v written TRANSPOSED [B,H,DK,S].
__global__ void __launch_bounds__(128)
qkv_proj(const __bf16* __restrict__ xb,
         const __bf16* __restrict__ wq, const __bf16* __restrict__ wk,
         const __bf16* __restrict__ wv,
         __bf16* __restrict__ qo, __bf16* __restrict__ ko, __bf16* __restrict__ vo) {
  const int lane = threadIdx.x;
  const int col  = lane & 15;
  const int half = lane >> 4;
  const int m0 = blockIdx.x * 32;
  const int n0 = (blockIdx.y * 4 + threadIdx.y) * 16;

  const __bf16* xrow0 = xb + (size_t)(m0 + col) * Dq;
  const __bf16* xrow1 = xb + (size_t)(m0 + 16 + col) * Dq;
  const __bf16* qrow  = wq + (size_t)(n0 + col) * Dq;
  const __bf16* krow  = wk + (size_t)(n0 + col) * Dq;
  const __bf16* vrow  = wv + (size_t)(n0 + col) * Dq;

  v8f aq0 = {}, ak0 = {}, av0 = {};
  v8f aq1 = {}, ak1 = {}, av1 = {};
  for (int ks = 0; ks < Dq; ks += 32) {
    v16bf a0 = load_afrag(xrow0, ks, half);
    v16bf a1 = load_afrag(xrow1, ks, half);
    const int kb = ks + 16 * half;
    v16bf bq = *(const v16bf*)(qrow + kb);
    v16bf bk = *(const v16bf*)(krow + kb);
    v16bf bv = *(const v16bf*)(vrow + kb);
    aq0 = wmma_bf16(a0, bq, aq0);  aq1 = wmma_bf16(a1, bq, aq1);
    ak0 = wmma_bf16(a0, bk, ak0);  ak1 = wmma_bf16(a1, bk, ak1);
    av0 = wmma_bf16(a0, bv, av0);  av1 = wmma_bf16(a1, bv, av1);
  }
  const int n  = n0 + col;
  const int h  = n >> 6;    // n / DK
  const int dk = n & 63;
#pragma unroll
  for (int sub = 0; sub < 2; ++sub) {
    const v8f& rq = sub ? aq1 : aq0;
    const v8f& rk = sub ? ak1 : ak0;
    const v8f& rv = sub ? av1 : av0;
#pragma unroll
    for (int i = 0; i < 8; ++i) {
      const int tok = m0 + 16 * sub + i + 8 * half;
      const int b = tok >> 11;          // / S
      const int s = tok & 2047;
      const size_t bh = (size_t)(b * Hq + h);
      qo[(bh * Sq + s) * DKq + dk] = (__bf16)rq[i];
      ko[(bh * Sq + s) * DKq + dk] = (__bf16)rk[i];
      vo[(bh * DKq + dk) * Sq + s] = (__bf16)rv[i];   // transposed
    }
  }
}

// ---------------- kernel 3: causal flash attention ----------------
// One wave per (b, h, 16-query tile); the 4 waves of a block share (b,h) and
// consume the same key-blocks, staged once into double-buffered LDS via
// async loads.  Scores computed TRANSPOSED (S^T = K @ Q^T) so the D-fragments
// of two 16x16 score tiles concatenate exactly into the A-operand
// (16 queries x 32 keys) of the P@V WMMA.
__global__ void __launch_bounds__(128)
attn(const __bf16* __restrict__ qb, const __bf16* __restrict__ kbuf,
     const __bf16* __restrict__ vtb, __bf16* __restrict__ ao) {
  __shared__ __bf16 smem[2][BUFEL];

  const int lane = threadIdx.x;
  const int col  = lane & 15;
  const int half = lane >> 4;
  const int tid  = threadIdx.y * 32 + lane;      // 0..127
  const int wid = blockIdx.x * 4 + threadIdx.y;  // 0..8191
  const int qt  = wid & 127;                     // query tile (S/16)
  const int bh  = wid >> 7;                      // 0..63
  const int qbase = qt * 16;

  const __bf16* qp  = qb  + (size_t)bh * Sq * DKq;
  const __bf16* kp  = kbuf + (size_t)bh * Sq * DKq;
  const __bf16* vtp = vtb + (size_t)bh * DKq * Sq;   // [dk, s]

  // Q as B operand (d x q), reused for every key block. 2 K-steps over DK=64.
  v16bf bqf[2];
#pragma unroll
  for (int t = 0; t < 2; ++t) {
    const int d0 = t * 32 + 16 * half;
    bqf[t] = *(const v16bf*)(qp + (size_t)(qbase + col) * DKq + d0);
  }

  const float scale = 0.125f;   // 1/sqrt(DK)
  float m_run = -1e30f, l_run = 0.f;
  v8f o0 = {}, o1 = {}, o2 = {}, o3 = {};

  const int nb    = (qbase + 16 + 31) >> 5;              // my causal blocks
  const int qtmax = (blockIdx.x * 4 + 3) & 127;          // last qt in block
  const int nbmax = (qtmax * 16 + 16 + 31) >> 5;         // uniform loop count

  // ---- stage one 32-key block (K: 32x64, V^T: 64x32) into LDS buffer ----
  auto stage = [&](int bi, int j0) {
    __bf16* kd = &smem[bi][0];
    __bf16* vd = &smem[bi][KEL];
    // K: 256 chunks of 8 el; thread handles chunks tid and tid+128.
#pragma unroll
    for (int c = tid; c < 256; c += 128) {
      const int r = c >> 3, o = (c & 7) * 8;
      async_ld16(kd + r * KST + o, kp + (size_t)(j0 + r) * DKq + o);
    }
    // V^T: 256 chunks of 8 el (64 rows x 4 chunks).
#pragma unroll
    for (int c = tid; c < 256; c += 128) {
      const int r = c >> 2, o = (c & 3) * 8;
      async_ld16(vd + r * VST + o, vtp + (size_t)r * Sq + j0 + o);
    }
  };

  stage(0, 0);
  wait_async0();
  __syncthreads();

  for (int jb = 0; jb < nbmax; ++jb) {
    const int bi = jb & 1;
    if (jb + 1 < nbmax) stage(bi ^ 1, (jb + 1) * 32);   // async next block

    if (jb < nb) {
      const int j0 = jb * 32;
      const __bf16* kl = &smem[bi][0];
      const __bf16* vl = &smem[bi][KEL];

      v8f s0 = {}, s1 = {};   // S^T tiles: keys j0..j0+15 / j0+16..j0+31
#pragma unroll
      for (int t = 0; t < 2; ++t) {
        const int d0 = t * 32 + 16 * half;
        v16bf a0 = pack16(*(const v8bf*)(kl + col * KST + d0),
                          *(const v8bf*)(kl + col * KST + d0 + 8));
        v16bf a1 = pack16(*(const v8bf*)(kl + (16 + col) * KST + d0),
                          *(const v8bf*)(kl + (16 + col) * KST + d0 + 8));
        s0 = wmma_bf16(a0, bqf[t], s0);
        s1 = wmma_bf16(a1, bqf[t], s1);
      }
      // scale + causal mask. S^T D layout: elem i -> key j0 + i + 8*half
      // (and +16 for s1), query q = qbase + col.
      const int qg = qbase + col;
      float vmax = -1e30f;
#pragma unroll
      for (int i = 0; i < 8; ++i) {
        const int jg0 = j0 + i + 8 * half;
        float x0 = s0[i] * scale; if (jg0 > qg)      x0 = -1e30f;
        float x1 = s1[i] * scale; if (jg0 + 16 > qg) x1 = -1e30f;
        s0[i] = x0; s1[i] = x1;
        vmax = fmaxf(vmax, fmaxf(x0, x1));
      }
      vmax = fmaxf(vmax, __shfl_xor(vmax, 16, 32));
      const float m_new = fmaxf(m_run, vmax);
      const float alpha = __expf(m_run - m_new);
      float bsum = 0.f;
      v16bf ap;   // P fragment == A operand of P@V (16 q x 32 keys)
#pragma unroll
      for (int i = 0; i < 8; ++i) {
        const float p0 = __expf(s0[i] - m_new);
        const float p1 = __expf(s1[i] - m_new);
        bsum += p0 + p1;
        ap[i]     = (__bf16)p0;
        ap[i + 8] = (__bf16)p1;
      }
      bsum += __shfl_xor(bsum, 16, 32);
      l_run = l_run * alpha + bsum;
      m_run = m_new;
      float af[8];
#pragma unroll
      for (int i = 0; i < 8; ++i) af[i] = __shfl(alpha, i + 8 * half, 32);
#pragma unroll
      for (int i = 0; i < 8; ++i) {
        o0[i] *= af[i]; o1[i] *= af[i]; o2[i] *= af[i]; o3[i] *= af[i];
      }
      // P @ V: B operand from staged V^T, lane n = col, kk = 8*half + i, 16 + 8*half + i.
#pragma unroll
      for (int d = 0; d < 4; ++d) {
        v16bf bvf = pack16(*(const v8bf*)(vl + (d * 16 + col) * VST + 8 * half),
                           *(const v8bf*)(vl + (d * 16 + col) * VST + 8 * half + 16));
        if      (d == 0) o0 = wmma_bf16(ap, bvf, o0);
        else if (d == 1) o1 = wmma_bf16(ap, bvf, o1);
        else if (d == 2) o2 = wmma_bf16(ap, bvf, o2);
        else             o3 = wmma_bf16(ap, bvf, o3);
      }
    }
    wait_async0();     // next buffer fully in LDS
    __syncthreads();   // all waves done reading current buffer
  }

  // finalize: divide by row-sum, store to [token, D] bf16.
  const int b = bh >> 4;
  const int h = bh & 15;
  float rf[8];
#pragma unroll
  for (int i = 0; i < 8; ++i) {
    const float lr = __shfl(l_run, i + 8 * half, 32);
    rf[i] = 1.f / lr;
  }
#pragma unroll
  for (int i = 0; i < 8; ++i) {
    const int row = qbase + i + 8 * half;
    __bf16* dst = ao + (size_t)(b * Sq + row) * Dq + h * DKq + col;
    dst[0]  = (__bf16)(o0[i] * rf[i]);
    dst[16] = (__bf16)(o1[i] * rf[i]);
    dst[32] = (__bf16)(o2[i] * rf[i]);
    dst[48] = (__bf16)(o3[i] * rf[i]);
  }
}

// ---------------- kernel 4: output projection (f32 result) ----------------
// 32(M) x 32(N) per wave: 4 WMMAs per K-step, A and B fragments each reused.
__global__ void __launch_bounds__(128)
out_proj(const __bf16* __restrict__ ab, const __bf16* __restrict__ wo,
         float* __restrict__ out) {
  const int lane = threadIdx.x;
  const int col  = lane & 15;
  const int half = lane >> 4;
  const int m0 = blockIdx.x * 32;
  const int n0 = (blockIdx.y * 4 + threadIdx.y) * 32;
  const __bf16* arow0 = ab + (size_t)(m0 + col) * Dq;
  const __bf16* arow1 = ab + (size_t)(m0 + 16 + col) * Dq;
  const __bf16* wrow0 = wo + (size_t)(n0 + col) * Dq;
  const __bf16* wrow1 = wo + (size_t)(n0 + 16 + col) * Dq;
  v8f c00 = {}, c01 = {}, c10 = {}, c11 = {};
  for (int ks = 0; ks < Dq; ks += 32) {
    v16bf a0 = load_afrag(arow0, ks, half);
    v16bf a1 = load_afrag(arow1, ks, half);
    const int kb = ks + 16 * half;
    v16bf b0 = *(const v16bf*)(wrow0 + kb);
    v16bf b1 = *(const v16bf*)(wrow1 + kb);
    c00 = wmma_bf16(a0, b0, c00);  c01 = wmma_bf16(a0, b1, c01);
    c10 = wmma_bf16(a1, b0, c10);  c11 = wmma_bf16(a1, b1, c11);
  }
#pragma unroll
  for (int ms = 0; ms < 2; ++ms) {
#pragma unroll
    for (int ns = 0; ns < 2; ++ns) {
      const v8f& c = ms ? (ns ? c11 : c10) : (ns ? c01 : c00);
      const int n = n0 + 16 * ns + col;
#pragma unroll
      for (int i = 0; i < 8; ++i) {
        const int tok = m0 + 16 * ms + i + 8 * half;
        out[(size_t)tok * Dq + n] = c[i];
      }
    }
  }
}

// ---------------- launch ----------------
extern "C" void kernel_launch(void* const* d_in, const int* in_sizes, int n_in,
                              void* d_out, int out_size, void* d_ws, size_t ws_size,
                              hipStream_t stream) {
  const float* x  = (const float*)d_in[0];
  const float* Wq = (const float*)d_in[1];
  const float* Wk = (const float*)d_in[2];
  const float* Wv = (const float*)d_in[3];
  const float* Wo = (const float*)d_in[4];
  float* out = (float*)d_out;

  // Workspace layout (bf16 elements), total ~93 MB.
  const size_t nX = (size_t)NT * Dq;              // 8.39M: x, attn-out
  const size_t nW = (size_t)Dq * Dq;              // 1.05M per weight
  const size_t nH = (size_t)Bq * Hq * Sq * DKq;   // 8.39M per q/k/v
  __bf16* xb   = (__bf16*)d_ws;
  __bf16* wqb  = xb + nX;
  __bf16* wkb  = wqb + nW;
  __bf16* wvb  = wkb + nW;
  __bf16* wob  = wvb + nW;
  __bf16* qbuf = wob + nW;
  __bf16* kbuf = qbuf + nH;
  __bf16* vbuf = kbuf + nH;   // transposed [B,H,DK,S]
  __bf16* abuf = vbuf + nH;

  cvt_f32_bf16<<<2048, 256, 0, stream>>>(x,  xb,  (int)nX);
  cvt_f32_bf16<<<1024, 256, 0, stream>>>(Wq, wqb, (int)nW);
  cvt_f32_bf16<<<1024, 256, 0, stream>>>(Wk, wkb, (int)nW);
  cvt_f32_bf16<<<1024, 256, 0, stream>>>(Wv, wvb, (int)nW);
  cvt_f32_bf16<<<1024, 256, 0, stream>>>(Wo, wob, (int)nW);

  dim3 blk(32, 4);   // 4 wave32s per block
  qkv_proj<<<dim3(NT / 32, (Dq / 16) / 4), blk, 0, stream>>>(
      xb, wqb, wkb, wvb, qbuf, kbuf, vbuf);
  attn<<<dim3((Bq * Hq * (Sq / 16)) / 4), blk, 0, stream>>>(
      qbuf, kbuf, vbuf, abuf);
  out_proj<<<dim3(NT / 32, (Dq / 32) / 4), blk, 0, stream>>>(abuf, wob, out);
}
